// ConsciousTransformer_76871324664393
// MI455X (gfx1250) — matrix operand order repaired and checked by reading, not compiled, not run on hardware
//
#include <hip/hip_runtime.h>
#include <hip/hip_bf16.h>
#include <math.h>

// ---------------- model dimensions (fixed by the reference) ----------------
constexpr int CC  = 512;     // model dim
constexpr int VV  = 32000;   // vocab
constexpr int LLn = 2;       // layers
constexpr int PPn = 2;       // phases
constexpr int EEn = 8;       // experts
constexpr int HMM = 1024;    // expert hidden
constexpr int HHn = 8;       // heads
constexpr int HDn = 64;      // head dim
constexpr int MMn = 256;     // memory slots
constexpr int BBn = 2;
constexpr int TTn = 1024;
constexpr int BT  = BBn * TTn;   // 2048 tokens

typedef __attribute__((ext_vector_type(16))) __bf16 v16bf;
typedef __attribute__((ext_vector_type(8)))  float  v8f;
typedef unsigned int v4u __attribute__((ext_vector_type(4)));
typedef int v4i __attribute__((ext_vector_type(4)));
typedef int v8i __attribute__((ext_vector_type(8)));

__device__ __forceinline__ float siluf(float x) { return x / (1.0f + __expf(-x)); }
__device__ __forceinline__ float actf(float x, int act) {
  if (act == 1) return siluf(x);            // silu
  if (act == 2) return x * siluf(x);        // x * silu(x)  (MoE activation)
  if (act == 3) return tanhf(x);            // gate
  return x;
}

// ---------------------------------------------------------------------------
// TDM: async-load one 64-row x 32-element bf16 tile (row stride = rowstride
// elements) from global into LDS at byte offset lds_off.  LDS pad feature
// inserts 4 DWORDs after every 16 DWORDs (=1 row of 64B) -> 80B LDS row pitch
// (matches __shared__ [64][40] bf16 and is bank-conflict-free for b128 reads).
// D# layout per CDNA5 ISA.  This toolchain's builtin is
// the 6-arg form: (g0, g1, g2, g3, extra_group, cpol).
// ---------------------------------------------------------------------------
__device__ __forceinline__ void tdm_load_tile32x64(unsigned lds_off,
                                                   const __bf16* gsrc,
                                                   int rowstride)
{
  unsigned long long ga = (unsigned long long)(uintptr_t)gsrc;
  v4u g0;
  g0.x = 1u;                                   // count=1, user descriptor
  g0.y = lds_off;                              // lds_addr (bytes)
  g0.z = (unsigned)ga;                         // global_addr[31:0]
  g0.w = (unsigned)(ga >> 32) | 0x80000000u;   // global_addr[56:32] | type=2

  unsigned td0 = (unsigned)rowstride;          // tensor_dim0 (elements)
  unsigned td1 = 0x100000u;                    // tensor_dim1: large (no OOB)
  v8i g1;
  // data_size=1 (2B) | pad_enable | pad_interval=3 (16 dw) | pad_amount=3 (4 dw)
  g1[0] = (int)((1u << 16) | (1u << 20) | (3u << 22) | (3u << 25));
  g1[1] = (int)((td0 & 0xFFFFu) << 16);                    // tensor_dim0 lo
  g1[2] = (int)((td0 >> 16) | ((td1 & 0xFFFFu) << 16));    // td0 hi | td1 lo
  g1[3] = (int)((td1 >> 16) | (32u << 16));                // td1 hi | tile_dim0=32
  g1[4] = 64;                                              // tile_dim1=64, tile_dim2=0
  g1[5] = (int)td0;                                        // tensor_dim0_stride lo
  g1[6] = 0;                                               // stride hi | dim1_stride lo
  g1[7] = 0;
  v4i gz4 = {0, 0, 0, 0};
  v8i gz8 = {0, 0, 0, 0, 0, 0, 0, 0};
  __builtin_amdgcn_tensor_load_to_lds(g0, g1, gz4, gz4, gz8, 0);
}

// ---------------------------------------------------------------------------
// WMMA bf16 GEMM:  C[M,N](ldc) = act( A[M,K] @ Bt[N,K]^T + bias[N] )
// A is bf16 row-major [M,K]; Bt is the weight pre-transposed to [N,K] bf16 so
// BOTH operands are K-major (contiguous WMMA fragment reads, ds_load_b128).
// Tiles are streamed global->LDS by the Tensor Data Mover, double-buffered.
// Requires M%64==0, N%64==0, K%32==0 (true for every GEMM in this model).
// ---------------------------------------------------------------------------
__global__ __launch_bounds__(256)
void k_gemm_bf16(const __bf16* __restrict__ A, const __bf16* __restrict__ Bt,
                 const float* __restrict__ bias, float* __restrict__ C,
                 int M, int N, int K, int ldc, int act)
{
  __shared__ __bf16 As[2][64][40];   // 80B row pitch (TDM pad)
  __shared__ __bf16 Bs[2][64][40];

  const int tid  = threadIdx.x;
  const int lane = tid & 31;
  const int wave = tid >> 5;      // 0..7
  const int wm   = wave >> 2;     // 0..1  -> M strip of 32 rows
  const int wn   = wave & 3;      // 0..3  -> N strip of 16 cols
  const int tileM = blockIdx.y * 64;
  const int tileN = blockIdx.x * 64;

  const __bf16* Abase = A  + (size_t)tileM * K;
  const __bf16* Bbase = Bt + (size_t)tileN * K;

  const unsigned ldsA0 = (unsigned)(uintptr_t)&As[0][0][0];
  const unsigned ldsA1 = (unsigned)(uintptr_t)&As[1][0][0];
  const unsigned ldsB0 = (unsigned)(uintptr_t)&Bs[0][0][0];
  const unsigned ldsB1 = (unsigned)(uintptr_t)&Bs[1][0][0];

  v8f acc0 = {};
  v8f acc1 = {};
  const int nk = K >> 5;

  if (tid < 32) {                       // wave 0 drives the DMA
    tdm_load_tile32x64(ldsA0, Abase, K);
    tdm_load_tile32x64(ldsB0, Bbase, K);
  }

  int buf = 0;
  for (int ik = 0; ik < nk; ++ik) {
    if (tid < 32) __builtin_amdgcn_s_wait_tensorcnt(0);
    __syncthreads();                    // tile[buf] visible to all waves
    if (tid < 32 && (ik + 1) < nk) {    // prefetch next tile into other buffer
      int k0n = (ik + 1) << 5;
      tdm_load_tile32x64(buf ? ldsA0 : ldsA1, Abase + k0n, K);
      tdm_load_tile32x64(buf ? ldsB0 : ldsB1, Bbase + k0n, K);
    }

    // fragments per CDNA5 16-bit WMMA layouts; both reads are 2x16B contiguous
    const int ml  = lane & 15;
    const int kaB = (lane < 16) ? 0 : 8;    // A lane-half K base
    const int kbB = (lane < 16) ? 0 : 8;    // B lane-half K base
    v16bf a0, a1, b;
#pragma unroll
    for (int j = 0; j < 16; ++j) {
      int ka = kaB + ((j < 8) ? j : (j + 8));
      a0[j] = As[buf][wm * 32 + ml][ka];
      a1[j] = As[buf][wm * 32 + 16 + ml][ka];
      b[j]  = Bs[buf][wn * 16 + ml][kbB + ((j < 8) ? j : (j + 8))];
    }
    acc0 = __builtin_amdgcn_wmma_f32_16x16x32_bf16(false, a0, false, b, (short)0, acc0, false, false);
    acc1 = __builtin_amdgcn_wmma_f32_16x16x32_bf16(false, a1, false, b, (short)0, acc1, false, false);
    buf ^= 1;
  }

  // epilogue: C elem r -> row = r + (lane<16?0:8), col = lane&15
  const int colg  = tileN + wn * 16 + (lane & 15);
  const int rbase = (lane < 16) ? 0 : 8;
  const float bv  = bias ? bias[colg] : 0.0f;
#pragma unroll
  for (int r = 0; r < 8; ++r) {
    int rowg0 = tileM + wm * 32 + rbase + r;
    int rowg1 = rowg0 + 16;
    C[(size_t)rowg0 * ldc + colg] = actf(acc0[r] + bv, act);
    C[(size_t)rowg1 * ldc + colg] = actf(acc1[r] + bv, act);
  }
}

// ---------------------------------------------------------------------------
// Precision conversion kernels (run once per launch for weights)
// ---------------------------------------------------------------------------
__global__ void k_cvt_bf16(const float* __restrict__ src, __bf16* __restrict__ dst, int n)
{
  int i = blockIdx.x * blockDim.x + threadIdx.x;
  if (i < n) dst[i] = (__bf16)src[i];
}

// weight W[K,N] f32 (row-major) -> Wt[N,K] bf16 (row-major, K contiguous)
__global__ void k_wt_bf16(const float* __restrict__ w, __bf16* __restrict__ wt, int K, int N)
{
  int i = blockIdx.x * blockDim.x + threadIdx.x;
  if (i < K * N) {
    int n = i / K, k = i % K;
    wt[(size_t)n * K + k] = (__bf16)w[(size_t)k * N + n];
  }
}

// ---------------------------------------------------------------------------
// Misc kernels
// ---------------------------------------------------------------------------
__global__ __launch_bounds__(256)
void k_embed(const int* __restrict__ ids, const float* __restrict__ ew, float* __restrict__ x)
{
  int bt = blockIdx.x, tid = threadIdx.x;
  const float* src = ew + (size_t)ids[bt] * CC;
  float* dst = x + (size_t)bt * CC;
  dst[tid]       = src[tid];
  dst[tid + 256] = src[tid + 256];
}

__global__ __launch_bounds__(256)
void k_softmax_rows(float* __restrict__ buf, int cols)
{
  int row = blockIdx.x, tid = threadIdx.x;
  __shared__ float red[256];
  float* r = buf + (size_t)row * cols;
  float m = -1e30f;
  for (int i = tid; i < cols; i += 256) m = fmaxf(m, r[i]);
  red[tid] = m; __syncthreads();
  for (int st = 128; st; st >>= 1) { if (tid < st) red[tid] = fmaxf(red[tid], red[tid + st]); __syncthreads(); }
  m = red[0]; __syncthreads();
  float s = 0.f;
  for (int i = tid; i < cols; i += 256) { float e = __expf(r[i] - m); r[i] = e; s += e; }
  red[tid] = s; __syncthreads();
  for (int st = 128; st; st >>= 1) { if (tid < st) red[tid] += red[tid + st]; __syncthreads(); }
  float inv = 1.0f / red[0]; __syncthreads();
  for (int i = tid; i < cols; i += 256) r[i] *= inv;
}

__global__ void k_mean_tokens(const float* __restrict__ h, float* __restrict__ meanh)
{
  int idx = blockIdx.x * blockDim.x + threadIdx.x;   // B*C = 1024
  if (idx >= BBn * CC) return;
  int b = idx / CC, c = idx % CC;
  float s = 0.f;
  for (int t = 0; t < TTn; ++t) s += h[((size_t)(b * TTn + t)) * CC + c];
  meanh[idx] = s * (1.0f / TTn);
}

__global__ void k_aware(const float* __restrict__ meanh, const float* __restrict__ aww,
                        const float* __restrict__ awb, const float* __restrict__ dsc,
                        float* __restrict__ scl)
{
  int tid = threadIdx.x;
  if (tid < BBn * HHn) {
    int b = tid / HHn, hh = tid % HHn;
    float acc = awb[hh];
    for (int c = 0; c < CC; ++c) acc += meanh[b * CC + c] * aww[c * HHn + hh];
    scl[tid] = dsc[0] * acc;
  }
}

// one block per (b, head, t) row: scores -> softmax -> (+0.3*prior) -> attn@V -> silu
__global__ __launch_bounds__(256)
void k_attn(const float* __restrict__ qkv, const float* __restrict__ scale,
            float* attnbuf, int use_prior, int write_attn,
            float* __restrict__ hbuf, float* __restrict__ outsbuf, int phase)
{
  const int t = blockIdx.x, head = blockIdx.y, b = blockIdx.z;
  const int tid = threadIdx.x;
  __shared__ float qs[HDn];
  __shared__ float sc[TTn];
  __shared__ float red[256];

  const float* qrow = qkv + ((size_t)(b * TTn + t)) * (3 * CC) + head * HDn;
  if (tid < HDn) qs[tid] = qrow[tid];
  __syncthreads();

  const float scl = scale[b * HHn + head];
  float lmax = -1e30f;
  for (int s = tid; s < TTn; s += 256) {
    const float* krow = qkv + ((size_t)(b * TTn + s)) * (3 * CC) + CC + head * HDn;
    float acc = 0.f;
#pragma unroll 8
    for (int d = 0; d < HDn; ++d) acc += qs[d] * krow[d];
    acc *= scl;
    sc[s] = acc;
    lmax = fmaxf(lmax, acc);
  }
  red[tid] = lmax; __syncthreads();
  for (int st = 128; st; st >>= 1) { if (tid < st) red[tid] = fmaxf(red[tid], red[tid + st]); __syncthreads(); }
  float m = red[0]; __syncthreads();
  float lsum = 0.f;
  for (int s = tid; s < TTn; s += 256) { float e = __expf(sc[s] - m); sc[s] = e; lsum += e; }
  red[tid] = lsum; __syncthreads();
  for (int st = 128; st; st >>= 1) { if (tid < st) red[tid] += red[tid + st]; __syncthreads(); }
  float inv = 1.0f / red[0]; __syncthreads();

  const size_t arow = (((size_t)(b * HHn + head)) * TTn + t) * TTn;
  for (int s = tid; s < TTn; s += 256) {
    float a = sc[s] * inv;
    if (use_prior)  a += 0.3f * attnbuf[arow + s];
    if (write_attn) attnbuf[arow + s] = a;
    sc[s] = a;
  }
  __syncthreads();

  // o[d] = sum_s attn[s] * V[s,d]; 4 partial strips of 256
  const int d = tid & 63;
  const int part = tid >> 6;
  float acc = 0.f;
  for (int s = part * 256; s < part * 256 + 256; ++s) {
    const float* vrow = qkv + ((size_t)(b * TTn + s)) * (3 * CC) + 2 * CC + head * HDn;
    acc += sc[s] * vrow[d];
  }
  red[tid] = acc; __syncthreads();
  if (tid < HDn) {
    float o  = red[tid] + red[tid + 64] + red[tid + 128] + red[tid + 192];
    float hv = siluf(o);
    size_t bt = (size_t)(b * TTn + t);
    hbuf[bt * CC + head * HDn + tid] = hv;
    outsbuf[bt * (PPn * CC) + phase * CC + head * HDn + tid] = hv;
  }
}

__global__ __launch_bounds__(256)
void k_layernorm(const float* __restrict__ merged, const float* __restrict__ xin,
                 const float* __restrict__ g, const float* __restrict__ bb,
                 float* __restrict__ xn)
{
  int row = blockIdx.x, tid = threadIdx.x;
  __shared__ float rs[256], rq[256];
  size_t base = (size_t)row * CC;
  float v0 = merged[base + tid]       + xin[base + tid];
  float v1 = merged[base + tid + 256] + xin[base + tid + 256];
  rs[tid] = v0 + v1;
  rq[tid] = v0 * v0 + v1 * v1;
  __syncthreads();
  for (int st = 128; st; st >>= 1) {
    if (tid < st) { rs[tid] += rs[tid + st]; rq[tid] += rq[tid + st]; }
    __syncthreads();
  }
  float mu  = rs[0] * (1.0f / CC);
  float var = rq[0] * (1.0f / CC) - mu * mu;
  float is  = rsqrtf(var + 1e-5f);
  xn[base + tid]       = (v0 - mu) * is * g[tid]       + bb[tid];
  xn[base + tid + 256] = (v1 - mu) * is * g[tid + 256] + bb[tid + 256];
}

// per-token: scores = softmax(xn @ gw + gb); dense top-2 combine weights
__global__ void k_gate(const float* __restrict__ xn, const float* __restrict__ gw,
                       const float* __restrict__ gb, float* __restrict__ wgt)
{
  int row = blockIdx.x, tid = threadIdx.x;
  __shared__ float logits[EEn];
  if (tid < EEn) {
    float acc = gb[tid];
    const float* xr = xn + (size_t)row * CC;
    for (int c = 0; c < CC; ++c) acc += xr[c] * gw[c * EEn + tid];
    logits[tid] = acc;
  }
  __syncthreads();
  if (tid == 0) {
    float m = logits[0];
    for (int e = 1; e < EEn; ++e) m = fmaxf(m, logits[e]);
    float p[EEn], s = 0.f;
    for (int e = 0; e < EEn; ++e) { p[e] = __expf(logits[e] - m); s += p[e]; }
    for (int e = 0; e < EEn; ++e) p[e] /= s;
    int i1 = 0;
    for (int e = 1; e < EEn; ++e) if (p[e] > p[i1]) i1 = e;
    int i2 = -1;
    for (int e = 0; e < EEn; ++e) { if (e == i1) continue; if (i2 < 0 || p[e] > p[i2]) i2 = e; }
    for (int e = 0; e < EEn; ++e) wgt[(size_t)row * EEn + e] = (e == i1 || e == i2) ? p[e] : 0.f;
  }
}

__global__ __launch_bounds__(256)
void k_quality(const float* __restrict__ qhid, const float* __restrict__ q2w,
               const float* __restrict__ q2b, float* __restrict__ quality)
{
  int row = blockIdx.x, tid = threadIdx.x;
  __shared__ float red[256];
  const float* r = qhid + (size_t)row * (4 * CC);
  float acc = 0.f;
  for (int i = tid; i < 4 * CC; i += 256) acc += r[i] * q2w[i];
  red[tid] = acc; __syncthreads();
  for (int st = 128; st; st >>= 1) { if (tid < st) red[tid] += red[tid + st]; __syncthreads(); }
  if (tid == 0) quality[row] = 1.0f / (1.0f + __expf(-(red[0] + q2b[0])));
}

// ---- elementwise fusion kernels ----
__global__ void k_copy_cols(float* __restrict__ dst, int ldd, const float* __restrict__ src,
                            int lds_, int rows, int cols)
{
  int i = blockIdx.x * blockDim.x + threadIdx.x;
  if (i < rows * cols) {
    int r = i / cols, c = i % cols;
    dst[(size_t)r * ldd + c] = src[(size_t)r * lds_ + c];
  }
}

__global__ void k_axpy(float* __restrict__ dst, const float* __restrict__ a,
                       const float* __restrict__ b, float alpha, int n)
{
  int i = blockIdx.x * blockDim.x + threadIdx.x;
  if (i < n) dst[i] = a[i] + alpha * b[i];
}

__global__ void k_rowscale_1mq(float* __restrict__ dst, const float* __restrict__ xin,
                               const float* __restrict__ quality, int n)
{
  int i = blockIdx.x * blockDim.x + threadIdx.x;
  if (i < n) dst[i] = xin[i] * (1.0f - quality[i >> 9]);   // /CC
}

__global__ void k_moe_acc(float* __restrict__ moe, const float* __restrict__ ey,
                          const float* __restrict__ wgt, int e, int n)
{
  int i = blockIdx.x * blockDim.x + threadIdx.x;
  if (i < n) moe[i] += ey[i] * wgt[(size_t)(i >> 9) * EEn + e];
}

__global__ void k_combine(float* __restrict__ x, const float* __restrict__ reason,
                          const float* __restrict__ gate, const float* __restrict__ pscale,
                          const float* __restrict__ intro, int n)
{
  int i = blockIdx.x * blockDim.x + threadIdx.x;
  if (i < n) x[i] = reason[i] + gate[i] * pscale[i & (CC - 1)] * intro[i];
}

// ---------------------------------------------------------------------------
// Driver
// ---------------------------------------------------------------------------
extern "C" void kernel_launch(void* const* d_in, const int* in_sizes, int n_in,
                              void* d_out, int out_size, void* d_ws, size_t ws_size,
                              hipStream_t stream)
{
  (void)in_sizes; (void)n_in; (void)out_size; (void)ws_size;

  const int*   ids     = (const int*)  d_in[0];
  const float* embed_w = (const float*)d_in[1];
  const float* mem     = (const float*)d_in[2];
  const float* mp_w    = (const float*)d_in[3];
  const float* mp_b    = (const float*)d_in[4];
  const float* qkv_w   = (const float*)d_in[5];
  const float* qkv_b   = (const float*)d_in[6];
  const float* aw_w    = (const float*)d_in[7];
  const float* aw_b    = (const float*)d_in[8];
  const float* dscale  = (const float*)d_in[9];
  const float* mg_w    = (const float*)d_in[10];
  const float* mg_b    = (const float*)d_in[11];
  const float* ln_g    = (const float*)d_in[12];
  const float* ln_b    = (const float*)d_in[13];
  const float* gw      = (const float*)d_in[14];
  const float* gb      = (const float*)d_in[15];
  const float* ew1     = (const float*)d_in[16];
  const float* eb1     = (const float*)d_in[17];
  const float* ew2     = (const float*)d_in[18];
  const float* eb2     = (const float*)d_in[19];
  const float* q1w     = (const float*)d_in[20];
  const float* q1b     = (const float*)d_in[21];
  const float* q2w     = (const float*)d_in[22];
  const float* q2b     = (const float*)d_in[23];
  const float* errw    = (const float*)d_in[24];
  const float* errb    = (const float*)d_in[25];
  const float* ag1w    = (const float*)d_in[26];
  const float* ag1b    = (const float*)d_in[27];
  const float* ag2w    = (const float*)d_in[28];
  const float* ag2b    = (const float*)d_in[29];
  const float* pscale  = (const float*)d_in[30];
  const float* head_w  = (const float*)d_in[31];
  const float* head_b  = (const float*)d_in[32];
  float* out = (float*)d_out;

  // byte-granular workspace allocator (16B aligned slots)
  char* base = (char*)d_ws;
  size_t off = 0;
  auto allocF = [&](size_t n) { float*  p = (float*) (base + off); off += ((n * 4 + 15) & ~(size_t)15); return p; };
  auto allocH = [&](size_t n) { __bf16* p = (__bf16*)(base + off); off += ((n * 2 + 15) & ~(size_t)15); return p; };

  float* x       = allocF((size_t)BT * CC);
  float* xin     = allocF((size_t)BT * CC);
  float* h       = allocF((size_t)BT * CC);
  float* xcat    = allocF((size_t)BT * 2 * CC);
  float* en      = allocF((size_t)BT * MMn);
  float* qkv     = allocF((size_t)BT * 3 * CC);
  float* meanh   = allocF((size_t)BBn * CC);
  float* sclbuf  = allocF((size_t)BBn * HHn);
  float* attnbuf = allocF((size_t)BBn * HHn * TTn * TTn);
  float* outs    = allocF((size_t)BT * PPn * CC);
  float* merged  = allocF((size_t)BT * CC);
  float* xn      = allocF((size_t)BT * CC);
  float* wgt     = allocF((size_t)BT * EEn);
  float* eh      = allocF((size_t)BT * HMM);
  float* ey      = allocF((size_t)BT * CC);
  float* moe     = allocF((size_t)BT * CC);
  float* reason  = allocF((size_t)BT * CC);
  float* qhid    = allocF((size_t)BT * 4 * CC);
  float* quality = allocF((size_t)BT);
  float* tmp     = allocF((size_t)BT * CC);
  float* err     = allocF((size_t)BT * CC);
  float* intro   = allocF((size_t)BT * CC);
  float* gatebuf = allocF((size_t)BT * CC);

  // bf16 activation scratch + pre-transposed bf16 weights [N,K]
  __bf16* abf        = allocH((size_t)BT * 4 * CC);          // A operand (max K=2048)
  __bf16* ehbf       = allocH((size_t)BT * HMM);
  __bf16* wt_mem_en  = allocH((size_t)MMn * CC);             // en:  B=mem^T -> Bt=mem
  __bf16* wt_mem_ret = allocH((size_t)CC * MMn);             // ret: B=mem   -> Bt=mem^T
  __bf16* wt_mp      = allocH((size_t)CC * 2 * CC);
  __bf16* wt_qkv     = allocH((size_t)LLn * PPn * 3 * CC * CC);
  __bf16* wt_mg      = allocH((size_t)LLn * CC * PPn * CC);
  __bf16* wt_ew1     = allocH((size_t)LLn * EEn * HMM * CC);
  __bf16* wt_ew2     = allocH((size_t)LLn * EEn * CC * HMM);
  __bf16* wt_q1      = allocH((size_t)LLn * 4 * CC * CC);
  __bf16* wt_err     = allocH((size_t)LLn * CC * CC);
  __bf16* wt_ag1     = allocH((size_t)4 * CC * 2 * CC);
  __bf16* wt_ag2     = allocH((size_t)CC * 4 * CC);
  __bf16* wt_head    = allocH((size_t)VV * CC);

  const int NBT = BT * CC;                    // 2048*512 elementwise size
  const int EWG = (NBT + 255) / 256;
  auto nb = [](size_t n) { return (unsigned)((n + 255) / 256); };

  auto gemmH = [&](const __bf16* Abf, const __bf16* Bt, const float* bias, float* Cd,
                   int M, int N, int K, int ldc, int act) {
    dim3 g(N / 64, M / 64), bl(256);
    k_gemm_bf16<<<g, bl, 0, stream>>>(Abf, Bt, bias, Cd, M, N, K, ldc, act);
  };
  auto gemmF = [&](const float* A, const __bf16* Bt, const float* bias, float* Cd,
                   int M, int N, int K, int ldc, int act) {
    k_cvt_bf16<<<nb((size_t)M * K), 256, 0, stream>>>(A, abf, M * K);
    gemmH(abf, Bt, bias, Cd, M, N, K, ldc, act);
  };

  // ---- one-time weight conversion / transposition to bf16 [N,K] ----
  k_cvt_bf16<<<nb((size_t)MMn * CC), 256, 0, stream>>>(mem, wt_mem_en, MMn * CC);
  k_wt_bf16<<<nb((size_t)MMn * CC), 256, 0, stream>>>(mem, wt_mem_ret, MMn, CC);
  k_wt_bf16<<<nb((size_t)2 * CC * CC), 256, 0, stream>>>(mp_w, wt_mp, 2 * CC, CC);
  for (int lp = 0; lp < LLn * PPn; ++lp)
    k_wt_bf16<<<nb((size_t)CC * 3 * CC), 256, 0, stream>>>(
        qkv_w + (size_t)lp * CC * 3 * CC, wt_qkv + (size_t)lp * 3 * CC * CC, CC, 3 * CC);
  for (int l = 0; l < LLn; ++l) {
    k_wt_bf16<<<nb((size_t)PPn * CC * CC), 256, 0, stream>>>(
        mg_w + (size_t)l * PPn * CC * CC, wt_mg + (size_t)l * CC * PPn * CC, PPn * CC, CC);
    k_wt_bf16<<<nb((size_t)CC * 4 * CC), 256, 0, stream>>>(
        q1w + (size_t)l * CC * 4 * CC, wt_q1 + (size_t)l * 4 * CC * CC, CC, 4 * CC);
    k_wt_bf16<<<nb((size_t)CC * CC), 256, 0, stream>>>(
        errw + (size_t)l * CC * CC, wt_err + (size_t)l * CC * CC, CC, CC);
  }
  for (int le = 0; le < LLn * EEn; ++le) {
    k_wt_bf16<<<nb((size_t)CC * HMM), 256, 0, stream>>>(
        ew1 + (size_t)le * CC * HMM, wt_ew1 + (size_t)le * HMM * CC, CC, HMM);
    k_wt_bf16<<<nb((size_t)HMM * CC), 256, 0, stream>>>(
        ew2 + (size_t)le * HMM * CC, wt_ew2 + (size_t)le * CC * HMM, HMM, CC);
  }
  k_wt_bf16<<<nb((size_t)2 * CC * 4 * CC), 256, 0, stream>>>(ag1w, wt_ag1, 2 * CC, 4 * CC);
  k_wt_bf16<<<nb((size_t)4 * CC * CC), 256, 0, stream>>>(ag2w, wt_ag2, 4 * CC, CC);
  k_wt_bf16<<<nb((size_t)CC * VV), 256, 0, stream>>>(head_w, wt_head, CC, VV);

  // ---- embedding + memory bank ----
  k_embed<<<BT, 256, 0, stream>>>(ids, embed_w, x);
  k_copy_cols<<<EWG, 256, 0, stream>>>(xcat, 2 * CC, x, CC, BT, CC);
  gemmF(x, wt_mem_en, nullptr, en, BT, MMn, CC, MMn, 0);            // en = x @ mem^T
  k_softmax_rows<<<BT, 256, 0, stream>>>(en, MMn);
  gemmF(en, wt_mem_ret, nullptr, xcat + CC, BT, CC, MMn, 2 * CC, 0); // ret -> xcat[:,C:]
  gemmF(xcat, wt_mp, mp_b, x, BT, CC, 2 * CC, CC, 0);               // memory_proj

  for (int l = 0; l < LLn; ++l) {
    (void)hipMemcpyAsync(xin, x, (size_t)NBT * 4, hipMemcpyDeviceToDevice, stream);
    (void)hipMemcpyAsync(h,   x, (size_t)NBT * 4, hipMemcpyDeviceToDevice, stream);

    // ---- iterative reasoning phases ----
    for (int p = 0; p < PPn; ++p) {
      int lp = l * PPn + p;
      gemmF(h, wt_qkv + (size_t)lp * 3 * CC * CC, qkv_b + (size_t)lp * 3 * CC,
            qkv, BT, 3 * CC, CC, 3 * CC, 0);
      k_mean_tokens<<<(BBn * CC + 255) / 256, 256, 0, stream>>>(h, meanh);
      k_aware<<<1, 32, 0, stream>>>(meanh, aw_w + (size_t)lp * CC * HHn,
                                    aw_b + (size_t)lp * HHn, dscale + lp, sclbuf);
      dim3 ag(TTn, HHn, BBn);
      k_attn<<<ag, 256, 0, stream>>>(qkv, sclbuf, attnbuf,
                                     /*use_prior=*/(p > 0), /*write_attn=*/(p < PPn - 1),
                                     h, outs, p);
    }
    gemmF(outs, wt_mg + (size_t)l * CC * PPn * CC, mg_b + (size_t)l * CC,
          merged, BT, CC, PPn * CC, CC, 0);
    k_layernorm<<<BT, 256, 0, stream>>>(merged, xin, ln_g + (size_t)l * CC,
                                        ln_b + (size_t)l * CC, xn);

    // ---- MoE (dense-equivalent math: all experts, top-2 combine weights) ----
    k_gate<<<BT, 64, 0, stream>>>(xn, gw + (size_t)l * CC * EEn, gb + (size_t)l * EEn, wgt);
    (void)hipMemsetAsync(moe, 0, (size_t)NBT * 4, stream);
    k_cvt_bf16<<<EWG, 256, 0, stream>>>(xn, abf, NBT);              // xn bf16, reused 8x
    for (int e = 0; e < EEn; ++e) {
      size_t le = (size_t)(l * EEn + e);
      gemmH(abf, wt_ew1 + le * HMM * CC, eb1 + le * HMM, eh, BT, HMM, CC, HMM, /*x*silu*/2);
      k_cvt_bf16<<<nb((size_t)BT * HMM), 256, 0, stream>>>(eh, ehbf, BT * HMM);
      gemmH(ehbf, wt_ew2 + le * CC * HMM, eb2 + le * CC, ey, BT, CC, HMM, CC, 0);
      k_moe_acc<<<EWG, 256, 0, stream>>>(moe, ey, wgt, e, NBT);
    }
    k_axpy<<<EWG, 256, 0, stream>>>(reason, xn, moe, 0.5f, NBT);    // RATIO

    // ---- introspection ----
    gemmF(xin, wt_q1 + (size_t)l * 4 * CC * CC, q1b + (size_t)l * 4 * CC,
          qhid, BT, 4 * CC, CC, 4 * CC, /*silu*/1);
    k_quality<<<BT, 256, 0, stream>>>(qhid, q2w + (size_t)l * 4 * CC, q2b + l, quality);
    k_rowscale_1mq<<<EWG, 256, 0, stream>>>(tmp, xin, quality, NBT);
    gemmF(tmp, wt_err + (size_t)l * CC * CC, errb + (size_t)l * CC, err, BT, CC, CC, CC, 0);
    k_axpy<<<EWG, 256, 0, stream>>>(intro, xin, err, 0.3f, NBT);

    // ---- autonomous learning gate ----
    k_copy_cols<<<EWG, 256, 0, stream>>>(xcat, 2 * CC, reason, CC, BT, CC);
    k_copy_cols<<<EWG, 256, 0, stream>>>(xcat + CC, 2 * CC, intro, CC, BT, CC);
    gemmF(xcat, wt_ag1, ag1b, qhid, BT, 4 * CC, 2 * CC, 4 * CC, /*silu*/1);
    gemmF(qhid, wt_ag2, ag2b, gatebuf, BT, CC, 4 * CC, CC, /*tanh*/3);
    k_combine<<<EWG, 256, 0, stream>>>(x, reason, gatebuf, pscale, intro, NBT);
  }

  // ---- vocab head ----
  gemmF(x, wt_head, head_b, out, BT, VV, CC, VV, 0);
}
